// Attention_59236188946841
// MI455X (gfx1250) — matrix operand order repaired and checked by reading, not compiled, not run on hardware
//
#include <hip/hip_runtime.h>
#include <hip/hip_bf16.h>
#include <stdint.h>

// ---------------------------------------------------------------------------
// Attention: out = softmax((X W^T + b) @ Enc) @ Enc
//   X   [8*1024, 1024] f32      W [2048, 1024] f32   b [2048] f32
//   Enc [8, 2048, 2048] f32     out [8, 1024, 2048] f32
// bf16 WMMA operands, f32 accumulation, f32 softmax.
// Workspace: buf0 = 8192*2048 bf16 (32 MB, proj then attn)
//            buf1 = 8192*2048 f32  (64 MB, scores)        total 96 MB
// ---------------------------------------------------------------------------

typedef __bf16 bf16_t;
typedef bf16_t v16bf __attribute__((ext_vector_type(16)));
typedef bf16_t v8bf  __attribute__((ext_vector_type(8)));
typedef float  v8f   __attribute__((ext_vector_type(8)));
typedef float  v4f   __attribute__((ext_vector_type(4)));
typedef int    v4i_t __attribute__((ext_vector_type(4)));

#define BM 256          // block tile M (4 waves x 64)
#define BN 128          // block tile N (2 waves x 64)
#define BK 32           // K step (one bf16 WMMA K)
#define LDK 40          // padded LDS row stride in bf16 elems (80 B, 16B-aligned)

#if defined(__has_builtin)
# if __has_builtin(__builtin_amdgcn_global_load_async_to_lds_b128)
#  define HAVE_ASYNC_LDS 1
# endif
# if __has_builtin(__builtin_amdgcn_cvt_pk_bf16_f32)
#  define HAVE_CVT_PK_BF16 1
# endif
#endif
#ifndef HAVE_ASYNC_LDS
# define HAVE_ASYNC_LDS 0
#endif
#ifndef HAVE_CVT_PK_BF16
# define HAVE_CVT_PK_BF16 0
#endif

// two floats -> packed {bf16(b)[31:16], bf16(a)[15:0]}
__device__ __forceinline__ uint32_t pack2bf(float a, float b) {
#if HAVE_CVT_PK_BF16
  auto p = __builtin_amdgcn_cvt_pk_bf16_f32(a, b);   // v_cvt_pk_bf16_f32
  return __builtin_bit_cast(uint32_t, p);
#else
  union { float f; uint32_t u; } x, y; x.f = a; y.f = b;
  uint32_t ra = (x.u + 0x7FFFu + ((x.u >> 16) & 1u)) >> 16;
  uint32_t rb = (y.u + 0x7FFFu + ((y.u >> 16) & 1u)) & 0xFFFF0000u;
  return ra | rb;
#endif
}

__device__ __forceinline__ bf16_t f2bf(float f) {
#if HAVE_CVT_PK_BF16
  uint32_t u = pack2bf(f, 0.0f);
  return __builtin_bit_cast(bf16_t, (uint16_t)(u & 0xFFFFu));
#else
  union { float f; uint32_t u; } v; v.f = f;
  uint32_t r = v.u + 0x7FFFu + ((v.u >> 16) & 1u);   // round-to-nearest-even
  return __builtin_bit_cast(bf16_t, (uint16_t)(r >> 16));
#endif
}

// A fragment 16x32 bf16 from LDS tile As[.][LDK] (row-major M x K).
// lane<16: row M=lane, K = 0..7 | 16..23 ; lane>=16: row M=lane-16, K = 8..15 | 24..31
__device__ __forceinline__ v16bf load_frag_A(const bf16_t* As, int rowBase, int lane) {
  const int half = (lane >> 4) & 1;
  const int mr   = lane & 15;
  const bf16_t* p = As + (rowBase + mr) * LDK + half * 8;
  v8bf lo = *(const v8bf*)(p);
  v8bf hi = *(const v8bf*)(p + 16);
  v16bf r;
#pragma unroll
  for (int i = 0; i < 8; ++i) { r[i] = lo[i]; r[i + 8] = hi[i]; }
  return r;
}

// B fragment 32x16 bf16 from LDS tile Bs[.][LDK] (N-major: Bs[n][k]).
// lane<16: column N=lane, K = 0..7 | 16..23 ; lane>=16: column N=lane-16, K = 8..15 | 24..31
__device__ __forceinline__ v16bf load_frag_B(const bf16_t* Bs, int colBase, int lane) {
  const int half = (lane >> 4) & 1;
  const int nn   = lane & 15;
  const bf16_t* p = Bs + (colBase + nn) * LDK + half * 8;
  v8bf lo = *(const v8bf*)(p);
  v8bf hi = *(const v8bf*)(p + 16);
  v16bf r;
#pragma unroll
  for (int i = 0; i < 8; ++i) { r[i] = lo[i]; r[i + 8] = hi[i]; }
  return r;
}

// ---------------------------------------------------------------------------
// Kernel 1: proj = X @ W^T + b    -> bf16
// ---------------------------------------------------------------------------
__global__ __launch_bounds__(256)
void gemm_proj(const float* __restrict__ X, const float* __restrict__ W,
               const float* __restrict__ bias, bf16_t* __restrict__ P,
               int K, int N) {
  __shared__ bf16_t As[BM * LDK];
  __shared__ bf16_t Bs[BN * LDK];

  const int tid   = threadIdx.x;
  const int lane  = tid & 31;
  const int wid   = tid >> 5;
  const int waveM = wid & 3;       // 4 strips of 64 rows
  const int waveN = wid >> 2;      // 2 strips of 64 cols
  const int row0  = blockIdx.y * BM;
  const int n0    = blockIdx.x * BN;

  // hoisted staging addresses (advance by BK per iteration)
  const float* aSrc[8]; uint32_t* aDst[8];
#pragma unroll
  for (int c = 0; c < 8; ++c) {
    int idx = tid + c * 256;                // 2048 float4 chunks
    int r = idx >> 3, ck = (idx & 7) * 4;
    aSrc[c] = X + (uint32_t)(row0 + r) * (uint32_t)K + ck;
    aDst[c] = (uint32_t*)(As + r * LDK + ck);   // 8B aligned
  }
  const float* bSrc[4]; uint32_t* bDst[4];
#pragma unroll
  for (int c = 0; c < 4; ++c) {
    int idx = tid + c * 256;                // 1024 float4 chunks
    int r = idx >> 3, ck = (idx & 7) * 4;
    bSrc[c] = W + (uint32_t)(n0 + r) * (uint32_t)K + ck;
    bDst[c] = (uint32_t*)(Bs + r * LDK + ck);
  }

  v8f acc[4][4];
#pragma unroll
  for (int i = 0; i < 4; ++i)
#pragma unroll
    for (int j = 0; j < 4; ++j)
#pragma unroll
      for (int e = 0; e < 8; ++e) acc[i][j][e] = 0.0f;

  for (int k0 = 0; k0 < K; k0 += BK) {
#pragma unroll
    for (int c = 0; c < 8; ++c) {
      v4f xv = *(const v4f*)aSrc[c]; aSrc[c] += BK;
      aDst[c][0] = pack2bf(xv[0], xv[1]);
      aDst[c][1] = pack2bf(xv[2], xv[3]);
    }
#pragma unroll
    for (int c = 0; c < 4; ++c) {
      v4f wv = *(const v4f*)bSrc[c]; bSrc[c] += BK;
      bDst[c][0] = pack2bf(wv[0], wv[1]);
      bDst[c][1] = pack2bf(wv[2], wv[3]);
    }
    __syncthreads();

    v16bf aF[4], bF[4];
#pragma unroll
    for (int i = 0; i < 4; ++i) aF[i] = load_frag_A(As, waveM * 64 + i * 16, lane);
#pragma unroll
    for (int j = 0; j < 4; ++j) bF[j] = load_frag_B(Bs, waveN * 64 + j * 16, lane);
#pragma unroll
    for (int i = 0; i < 4; ++i)
#pragma unroll
      for (int j = 0; j < 4; ++j)
        acc[i][j] = __builtin_amdgcn_wmma_f32_16x16x32_bf16(
            false, aF[i], false, bF[j], (short)0, acc[i][j], false, false);
    __syncthreads();
  }

  // epilogue: 32-bit offset math (max flat index 16.7M < 2^31)
  const int half = lane >> 4;
  const int nn   = lane & 15;
  const int rowStart = row0 + waveM * 64 + half * 8;
  const int colStart = n0 + waveN * 64 + nn;
#pragma unroll
  for (int j = 0; j < 4; ++j) {
    const float bv = bias[colStart + j * 16];
#pragma unroll
    for (int i = 0; i < 4; ++i) {
      uint32_t off = (uint32_t)(rowStart + i * 16) * (uint32_t)N + colStart + j * 16;
      bf16_t* pp = P + off;
#pragma unroll
      for (int e = 0; e < 8; ++e) { *pp = f2bf(acc[i][j][e] + bv); pp += N; }
    }
  }
}

// ---------------------------------------------------------------------------
// Kernel 2/4: C = A(bf16) @ EncB(f32, [K][N] per batch)   -> f32
// A tile copied via CDNA5 async DMA to LDS; Enc staged transposed (bf16).
// ---------------------------------------------------------------------------
__global__ __launch_bounds__(256)
void gemm_bf16_enc(const bf16_t* __restrict__ A, const float* __restrict__ Enc,
                   float* __restrict__ C, int K, int N, int Mb) {
  __shared__ bf16_t As[BM * LDK];
  __shared__ bf16_t Bs[BN * LDK];

  const int tid   = threadIdx.x;
  const int lane  = tid & 31;
  const int wid   = tid >> 5;
  const int waveM = wid & 3;
  const int waveN = wid >> 2;
  const int z     = blockIdx.z;
  const int row0  = z * Mb + blockIdx.y * BM;     // < 8192
  const int n0    = blockIdx.x * BN;
  const float* encB = Enc + (size_t)z * (size_t)K * (size_t)N;

  const bf16_t* aSrc[4]; bf16_t* aDst[4];
#pragma unroll
  for (int c = 0; c < 4; ++c) {
    int idx = tid + c * 256;                // 1024 16B chunks (256 rows x 4)
    int r = idx >> 2, ck = (idx & 3) * 8;
    aSrc[c] = A + (uint32_t)(row0 + r) * (uint32_t)K + ck;
    aDst[c] = As + r * LDK + ck;
  }
  const float* bSrc[4]; bf16_t* bDst[4];
#pragma unroll
  for (int c = 0; c < 4; ++c) {
    int idx = tid + c * 256;                // 32 k-rows x 32 float4 cols
    int kk = idx >> 5, nn4 = (idx & 31) * 4;
    bSrc[c] = encB + (uint32_t)kk * (uint32_t)N + n0 + nn4;
    bDst[c] = Bs + nn4 * LDK + kk;          // transpose: store N-major
  }

  v8f acc[4][4];
#pragma unroll
  for (int i = 0; i < 4; ++i)
#pragma unroll
    for (int j = 0; j < 4; ++j)
#pragma unroll
      for (int e = 0; e < 8; ++e) acc[i][j][e] = 0.0f;

  const int encStep = BK * N;                // 65536 floats, fits int
  for (int k0 = 0; k0 < K; k0 += BK) {
    // A tile: pure bf16 copy, 16B per chunk (CDNA5 async DMA to LDS)
#pragma unroll
    for (int c = 0; c < 4; ++c) {
#if HAVE_ASYNC_LDS
      __builtin_amdgcn_global_load_async_to_lds_b128(
          (v4i_t*)aSrc[c], (v4i_t*)aDst[c], 0, 0);
#else
      *(v8bf*)aDst[c] = *(const v8bf*)aSrc[c];
#endif
      aSrc[c] += BK;
    }
    // Enc tile: f32 -> bf16, transposed into N-major LDS
#pragma unroll
    for (int c = 0; c < 4; ++c) {
      v4f ev = *(const v4f*)bSrc[c];
      if (k0 + BK < K) __builtin_prefetch(bSrc[c] + encStep, 0, 3);
      bSrc[c] += encStep;
#pragma unroll
      for (int j = 0; j < 4; ++j) bDst[c][j * LDK] = f2bf(ev[j]);
    }
#if HAVE_ASYNC_LDS
    asm volatile("s_wait_asynccnt 0" ::: "memory");
#endif
    __syncthreads();

    v16bf aF[4], bF[4];
#pragma unroll
    for (int i = 0; i < 4; ++i) aF[i] = load_frag_A(As, waveM * 64 + i * 16, lane);
#pragma unroll
    for (int j = 0; j < 4; ++j) bF[j] = load_frag_B(Bs, waveN * 64 + j * 16, lane);
#pragma unroll
    for (int i = 0; i < 4; ++i)
#pragma unroll
      for (int j = 0; j < 4; ++j)
        acc[i][j] = __builtin_amdgcn_wmma_f32_16x16x32_bf16(
            false, aF[i], false, bF[j], (short)0, acc[i][j], false, false);
    __syncthreads();
  }

  const int half = lane >> 4;
  const int nn   = lane & 15;
  const int rowStart = row0 + waveM * 64 + half * 8;
  const int colStart = n0 + waveN * 64 + nn;
#pragma unroll
  for (int i = 0; i < 4; ++i) {
#pragma unroll
    for (int j = 0; j < 4; ++j) {
      uint32_t off = (uint32_t)(rowStart + i * 16) * (uint32_t)N + colStart + j * 16;
      float* cp = C + off;
#pragma unroll
      for (int e = 0; e < 8; ++e) { *cp = acc[i][j][e]; cp += N; }
    }
  }
}

// ---------------------------------------------------------------------------
// Kernel 3: row softmax, f32 scores -> bf16 attn. One block per row (N=2048).
// ---------------------------------------------------------------------------
__global__ __launch_bounds__(256)
void softmax_rows(const float* __restrict__ S, bf16_t* __restrict__ Aout, int N) {
  __shared__ float red[256];
  const long row = blockIdx.x;
  const int tid = threadIdx.x;
  const float* sr = S + row * (long)N;

  v4f vals[2];
  float m = -3.402823466e+38f;
#pragma unroll
  for (int c = 0; c < 2; ++c) {
    vals[c] = *(const v4f*)(sr + tid * 4 + c * 1024);
#pragma unroll
    for (int i = 0; i < 4; ++i) m = fmaxf(m, vals[c][i]);
  }
  red[tid] = m;
  __syncthreads();
  for (int s = 128; s > 0; s >>= 1) {
    if (tid < s) red[tid] = fmaxf(red[tid], red[tid + s]);
    __syncthreads();
  }
  m = red[0];
  __syncthreads();

  float sum = 0.0f;
#pragma unroll
  for (int c = 0; c < 2; ++c)
#pragma unroll
    for (int i = 0; i < 4; ++i) {
      vals[c][i] = __expf(vals[c][i] - m);
      sum += vals[c][i];
    }
  red[tid] = sum;
  __syncthreads();
  for (int s = 128; s > 0; s >>= 1) {
    if (tid < s) red[tid] += red[tid + s];
    __syncthreads();
  }
  const float inv = 1.0f / red[0];
#pragma unroll
  for (int c = 0; c < 2; ++c) {
    uint32_t lo = pack2bf(vals[c][0] * inv, vals[c][1] * inv);
    uint32_t hi = pack2bf(vals[c][2] * inv, vals[c][3] * inv);
    uint32_t* op = (uint32_t*)(Aout + row * (long)N + tid * 4 + c * 1024);
    op[0] = lo; op[1] = hi;
  }
}

// ---------------------------------------------------------------------------
extern "C" void kernel_launch(void* const* d_in, const int* in_sizes, int n_in,
                              void* d_out, int out_size, void* d_ws, size_t ws_size,
                              hipStream_t stream) {
  (void)in_sizes; (void)n_in; (void)out_size;
  const int B = 8, TQ = 1024, H = 1024, TWO_H = 2048, S = 2048;
  const long ROWS = (long)B * TQ;                 // 8192

  const float* inp  = (const float*)d_in[0];      // [B, TQ, H]
  const float* enc  = (const float*)d_in[1];      // [B, S, S]
  const float* W    = (const float*)d_in[2];      // [2H, H]
  const float* bias = (const float*)d_in[3];      // [2H]
  float* out = (float*)d_out;                     // [B, TQ, S]

  const size_t buf0_bytes = (size_t)ROWS * TWO_H * sizeof(bf16_t);  // 32 MB
  const size_t buf1_bytes = (size_t)ROWS * S * sizeof(float);       // 64 MB
  if (ws_size < buf0_bytes + buf1_bytes) return;  // need 96 MB scratch

  bf16_t* buf0 = (bf16_t*)d_ws;                         // proj, then attn (bf16)
  float*  buf1 = (float*)((char*)d_ws + buf0_bytes);    // scores (f32)

  // 1) proj = X @ W^T + b  -> bf16
  gemm_proj<<<dim3(TWO_H / BN, ROWS / BM, 1), 256, 0, stream>>>(
      inp, W, bias, buf0, H, TWO_H);

  // 2) scores = proj @ enc -> f32
  gemm_bf16_enc<<<dim3(S / BN, TQ / BM, B), 256, 0, stream>>>(
      buf0, enc, buf1, TWO_H, S, TQ);

  // 3) attn = softmax(scores) -> bf16 (overwrites proj)
  softmax_rows<<<dim3((unsigned)ROWS), 256, 0, stream>>>(buf1, buf0, S);

  // 4) out = attn @ enc -> f32
  gemm_bf16_enc<<<dim3(S / BN, TQ / BM, B), 256, 0, stream>>>(
      buf0, enc, out, S, S, TQ);
}
